// VectorQuantizer_60284160967223
// MI455X (gfx1250) — matrix operand order repaired and checked by reading, not compiled, not run on hardware
//
#include <hip/hip_runtime.h>

// ---------------------------------------------------------------------------
// VQ assignment = GEMM (xc = X @ C^T) + argmin(c2 - 2*xc) + gather.
// bf16 WMMA (v_wmma_f32_16x16x32_bf16), wave32, 8 waves / 256-thread block.
// Register-blocked: each wave computes 32 tokens x 16 codes (2 WMMA tiles)
// per B-fragment load -> 1:1 wmma:ds_load_b128 ratio.
// ---------------------------------------------------------------------------

typedef __attribute__((ext_vector_type(16))) __bf16 v16bf;
typedef __attribute__((ext_vector_type(8)))  float  v8f;

constexpr int DIMS   = 256;     // D
constexpr int KCODES = 1024;    // K
constexpr int BATCH  = 16;      // B
constexpr int TSEQ   = 4096;    // T
constexpr int MT     = 64;      // tokens per workgroup
constexpr int NT     = 64;      // codes per chunk (4 wave-columns of 16)
constexpr int LDA    = DIMS + 16;  // padded LDS row stride (ushorts): 544B, 16B-aligned
constexpr int LDB    = DIMS + 16;

__device__ __forceinline__ unsigned short f2bf(float f) {
    unsigned u = __float_as_uint(f);
    unsigned r = u + 0x7FFFu + ((u >> 16) & 1u);   // round-to-nearest-even
    return (unsigned short)(r >> 16);
}

// --- prep: codebook f32 -> bf16 (ws) and c2[k] = ||c_k||^2 ------------------
__global__ __launch_bounds__(256) void vq_prep(const float* __restrict__ cb,
                                               unsigned short* __restrict__ cbbf,
                                               float* __restrict__ c2) {
    const int k = blockIdx.x;     // one code per block
    const int t = threadIdx.x;    // DIMS == blockDim.x == 256
    float v = cb[k * DIMS + t];
    cbbf[k * DIMS + t] = f2bf(v);
    __shared__ float red[256];
    red[t] = v * v;
    __syncthreads();
    for (int s = 128; s > 0; s >>= 1) {
        if (t < s) red[t] += red[t + s];
        __syncthreads();
    }
    if (t == 0) c2[k] = red[0];
}

// --- main: distances via WMMA, per-token argmin, gather ---------------------
__global__ __launch_bounds__(256) void vq_main(const float* __restrict__ x,
                                               const float* __restrict__ cbf32,
                                               const unsigned short* __restrict__ cbbf,
                                               const float* __restrict__ c2g,
                                               float* __restrict__ out) {
    // One 34816B buffer, time-multiplexed:
    //   phase 1: A staging  [MT][LDA] bf16 (34816B)
    //   phase 2: B tile     [NT][LDB] bf16 (34816B)
    //   phase 3: score/idx  [MT][NT] f32 + [MT][NT] i32 (32768B)
    __shared__ __align__(16) unsigned char smem[MT * LDA * 2 + KCODES * 4 + MT * 4];
    unsigned short* aS     = (unsigned short*)smem;
    unsigned short* bS     = (unsigned short*)smem;
    float*          scoreS = (float*)smem;                        // [MT][NT]
    int*            idxS   = (int*)(smem + MT * NT * 4);          // [MT][NT]
    float*          c2S    = (float*)(smem + MT * LDA * 2);
    int*            bestS  = (int*)(smem + MT * LDA * 2 + KCODES * 4);

    const int tid  = threadIdx.x;
    const int lane = tid & 31;
    const int wave = tid >> 5;        // 0..7
    const int wm   = wave >> 2;       // M super-tile 0..1 (32 tokens each, 2 WMMA tiles)
    const int wn   = wave & 3;        // N sub-tile 0..3 (16 codes each)
    const long tok0 = (long)blockIdx.x * MT;

    // ---- stage inputs f32 -> bf16 into LDS (coalesced float4 loads) ----
    {
        const float4* xin = (const float4*)(x + tok0 * DIMS);
        #pragma unroll
        for (int i = 0; i < 16; ++i) {
            int e    = tid + i * 256;          // 0..4095 float4 elements
            int trow = e >> 6;                 // DIMS/4 = 64 float4 per token
            int c4   = e & 63;
            float4 v = xin[e];
            unsigned p0 = (unsigned)f2bf(v.x) | ((unsigned)f2bf(v.y) << 16);
            unsigned p1 = (unsigned)f2bf(v.z) | ((unsigned)f2bf(v.w) << 16);
            uint2* dst = (uint2*)(aS + trow * LDA + c4 * 4);
            *dst = make_uint2(p0, p1);
        }
        for (int i = tid; i < KCODES; i += 256) c2S[i] = c2g[i];
    }
    __syncthreads();

    // ---- load A fragments once: 2 tiles x 8 k-chunks per wave (128 VGPRs) ----
    // ISA 16-bit A 16x32 layout: lanes 0-15 hold K {0..7,16..23}, lanes 16-31 {8..15,24..31}
    typedef union { uint4 u[2]; v16bf v; } frag_t;
    frag_t afr[2][8];
    #pragma unroll
    for (int s = 0; s < 2; ++s) {
        const int row = wm * 32 + s * 16 + (lane & 15);
        const int kb  = (lane < 16) ? 0 : 8;
        const unsigned short* ap = aS + row * LDA;
        #pragma unroll
        for (int kc = 0; kc < 8; ++kc) {
            afr[s][kc].u[0] = *(const uint4*)(ap + kc * 32 + kb);
            afr[s][kc].u[1] = *(const uint4*)(ap + kc * 32 + kb + 16);
        }
    }
    __syncthreads();   // A staging buffer now free for B tiles

    float mins[2][8];
    int   mini[2][8];
    #pragma unroll
    for (int s = 0; s < 2; ++s)
        #pragma unroll
        for (int e = 0; e < 8; ++e) { mins[s][e] = 3.4e38f; mini[s][e] = 0; }

    const int nlocal = lane & 15;
    const int kbB    = (lane < 16) ? 0 : 8;   // B 32x16: lanes 0-15 K {0..7,16..23}, lanes 16-31 {8..15,24..31}

    // ---- loop over code chunks (16 chunks of 64 codes) ----
    for (int nc = 0; nc < KCODES / NT; ++nc) {
        // stage bf16 codebook chunk (NT x DIMS = 32KB) into LDS, coalesced uint4
        {
            const uint4* src = (const uint4*)(cbbf + (long)nc * NT * DIMS);
            if (nc + 1 < KCODES / NT)
                __builtin_prefetch(cbbf + (long)(nc + 1) * NT * DIMS + tid * 128, 0, 1);
            #pragma unroll
            for (int i = 0; i < 8; ++i) {
                int e    = tid + i * 256;      // 0..2047 uint4 elements
                int nrow = e >> 5;             // 32 uint4 per code row
                int c8   = e & 31;
                *(uint4*)(bS + nrow * LDB + c8 * 8) = src[e];
            }
        }
        __syncthreads();

        v8f acc0 = {}, acc1 = {};
        const unsigned short* bp = bS + (wn * 16 + nlocal) * LDB;
        #pragma unroll
        for (int kc = 0; kc < 8; ++kc) {
            frag_t bfr;
            bfr.u[0] = *(const uint4*)(bp + kc * 32 + kbB);
            bfr.u[1] = *(const uint4*)(bp + kc * 32 + kbB + 16);
            // two independent WMMAs sharing one B fragment (M register blocking)
            acc0 = __builtin_amdgcn_wmma_f32_16x16x32_bf16(
                false, afr[0][kc].v, false, bfr.v, (short)0, acc0, false, false);
            acc1 = __builtin_amdgcn_wmma_f32_16x16x32_bf16(
                false, afr[1][kc].v, false, bfr.v, (short)0, acc1, false, false);
        }

        // score = c2[n] - 2*xc ; running argmin (strict < keeps lowest index)
        const int   ncode = nc * NT + wn * 16 + nlocal;
        const float c2v   = c2S[ncode];
        #pragma unroll
        for (int e = 0; e < 8; ++e) {
            float s0 = c2v - 2.0f * acc0[e];
            if (s0 < mins[0][e]) { mins[0][e] = s0; mini[0][e] = ncode; }
            float s1 = c2v - 2.0f * acc1[e];
            if (s1 < mins[1][e]) { mins[1][e] = s1; mini[1][e] = ncode; }
        }
        __syncthreads();   // bS about to be overwritten (or aliased by scores)
    }

    // ---- cross-lane argmin via LDS (scores alias the dead B tile) ----
    {
        const int col = wn * 16 + nlocal;                  // 0..63
        #pragma unroll
        for (int s = 0; s < 2; ++s)
            #pragma unroll
            for (int e = 0; e < 8; ++e) {
                int m = wm * 32 + s * 16 + e + ((lane < 16) ? 0 : 8);  // f32 C/D layout
                scoreS[m * NT + col] = mins[s][e];
                idxS[m * NT + col]   = mini[s][e];
            }
    }
    __syncthreads();
    if (tid < MT) {
        float bs = 3.4e38f; int bi = KCODES;
        for (int c = 0; c < NT; ++c) {
            float s  = scoreS[tid * NT + c];
            int   ix = idxS[tid * NT + c];
            if (s < bs || (s == bs && ix < bi)) { bs = s; bi = ix; }
        }
        bestS[tid] = bi;
    }
    __syncthreads();

    // ---- gather winning codebook rows (f32, L2-hot), 4 threads / token ----
    {
        const int sub  = tid & 3;
        const int mtok = tid >> 2;                    // 0..63
        const int best = bestS[mtok];
        const float4* src = (const float4*)(cbf32 + (long)best * DIMS);
        float4*       dst = (float4*)(out + (tok0 + mtok) * DIMS);
        #pragma unroll
        for (int j = 0; j < 16; ++j)
            dst[sub + j * 4] = src[sub + j * 4];
    }
}

// ---------------------------------------------------------------------------
extern "C" void kernel_launch(void* const* d_in, const int* in_sizes, int n_in,
                              void* d_out, int out_size, void* d_ws, size_t ws_size,
                              hipStream_t stream) {
    const float* x  = (const float*)d_in[0];   // [B,T,D] f32
    const float* cb = (const float*)d_in[1];   // [K,D]  f32
    float*       out = (float*)d_out;          // [B,T,D] f32

    unsigned short* cbbf = (unsigned short*)d_ws;                       // 512 KB bf16 codebook
    float*          c2   = (float*)((char*)d_ws + (size_t)KCODES * DIMS * 2); // 4 KB norms

    vq_prep<<<KCODES, 256, 0, stream>>>(cb, cbbf, c2);

    const int ntok = BATCH * TSEQ;             // 65536
    vq_main<<<ntok / MT, 256, 0, stream>>>(x, cb, cbbf, c2, out);
}
